// SelfAttention_89060441850326
// MI455X (gfx1250) — compile-verified, matrix operand order checked by reading, not run on hardware
//
#include <hip/hip_runtime.h>
#include <hip/hip_bf16.h>

typedef __bf16 bf16;
typedef __attribute__((ext_vector_type(16))) __bf16 v16bf;
typedef __attribute__((ext_vector_type(8)))  __bf16 v8bf;
typedef __attribute__((ext_vector_type(8)))  float  v8f;

// ---------- helpers ----------
__device__ __forceinline__ bf16 f2bf(float f) {
    unsigned u = __builtin_bit_cast(unsigned, f);
    u += 0x7FFFu + ((u >> 16) & 1u);          // round-to-nearest-even
    unsigned short s = (unsigned short)(u >> 16);
    return __builtin_bit_cast(bf16, s);
}

// 16-bit A/B fragment for V_WMMA_*_16X16X32_BF16 (wave32):
// lane&15 selects the stored row (A: M-row, B: N-column when stored N-major),
// lane>>4 selects the k-half; element e -> k = (e<8 ? e : e+8) + 8*(lane>=16).
// Both 8-element runs are contiguous in memory -> two 16B loads.
__device__ __forceinline__ v16bf load_frag(const bf16* __restrict__ base,
                                           int row, int ld, int kbase, int lane) {
    const bf16* p = base + (size_t)row * ld + kbase + ((lane >> 4) << 3);
    v8bf lo = *reinterpret_cast<const v8bf*>(p);
    v8bf hi = *reinterpret_cast<const v8bf*>(p + 16);
    return __builtin_shufflevector(lo, hi, 0,1,2,3,4,5,6,7,8,9,10,11,12,13,14,15);
}

__device__ __forceinline__ v8f wmma_bf16(v16bf a, v16bf b, v8f c) {
    return __builtin_amdgcn_wmma_f32_16x16x32_bf16(false, a, false, b, (short)0, c,
                                                   false, false);
}

#define B_   8
#define C_   512
#define HW_  1024
#define NH_  8
#define HD_  64
#define G_   32

// ---------- K0: fp32 -> bf16 weight convert ----------
__global__ __launch_bounds__(256) void cvt_kernel(const float* __restrict__ src,
                                                  bf16* __restrict__ dst, int n) {
    int i = blockIdx.x * 256 + threadIdx.x;
    if (i < n) dst[i] = f2bf(src[i]);
}

// ---------- K1: GroupNorm -> position-major bf16 n_t[b][p][c] ----------
__global__ __launch_bounds__(256) void gn_kernel(const float* __restrict__ x,
                                                 const float* __restrict__ gw,
                                                 const float* __restrict__ gb,
                                                 bf16* __restrict__ n_t) {
    int bg = blockIdx.x;
    int bb = bg >> 5, g = bg & 31;          // 16 channels per group
    int tid = threadIdx.x;
    const float* xb = x + (size_t)bb * C_ * HW_ + (size_t)g * 16 * HW_;

    float vals[4][16];
    float s = 0.f, s2 = 0.f;
    #pragma unroll
    for (int i = 0; i < 4; ++i) {
        int p = tid + i * 256;
        #pragma unroll
        for (int c = 0; c < 16; ++c) {
            float v = xb[c * HW_ + p];
            vals[i][c] = v; s += v; s2 += v * v;
        }
    }
    __shared__ float red0[256], red1[256];
    red0[tid] = s; red1[tid] = s2;
    __syncthreads();
    for (int off = 128; off > 0; off >>= 1) {
        if (tid < off) { red0[tid] += red0[tid + off]; red1[tid] += red1[tid + off]; }
        __syncthreads();
    }
    float mu   = red0[0] * (1.f / 16384.f);
    float var  = red1[0] * (1.f / 16384.f) - mu * mu;
    float rstd = rsqrtf(var + 1e-5f);

    float ws[16], bs[16];
    #pragma unroll
    for (int c = 0; c < 16; ++c) {
        float wc = gw[g * 16 + c], bc = gb[g * 16 + c];
        ws[c] = wc * rstd; bs[c] = bc - mu * rstd * wc;
    }
    #pragma unroll
    for (int i = 0; i < 4; ++i) {
        int p = tid + i * 256;
        v8bf o0, o1;
        #pragma unroll
        for (int c = 0; c < 8; ++c) {
            o0[c] = f2bf(vals[i][c]     * ws[c]     + bs[c]);
            o1[c] = f2bf(vals[i][c + 8] * ws[c + 8] + bs[c + 8]);
        }
        bf16* dst = n_t + ((size_t)bb * HW_ + p) * C_ + g * 16;
        *reinterpret_cast<v8bf*>(dst)     = o0;
        *reinterpret_cast<v8bf*>(dst + 8) = o1;
    }
}

// ---------- K2: QKV GEMM  out[p][d] = n_t[p][:] . qkv_w[d][:] ----------
// Scatter into q_t[b][h][p][64] (pre-scaled), k_t[b][h][p][64], v_cm[b][h][64][p].
__global__ __launch_bounds__(256) void qkv_kernel(const bf16* __restrict__ n_t,
                                                  const bf16* __restrict__ wq,
                                                  bf16* __restrict__ q_t,
                                                  bf16* __restrict__ k_t,
                                                  bf16* __restrict__ v_cm) {
    int bb = blockIdx.z;
    int Mb = blockIdx.x * 32;            // position base
    int Nb = blockIdx.y * 256;           // d base
    int tid = threadIdx.x, wave = tid >> 5, lane = tid & 31;
    int m0 = Mb + 16 * (wave & 1);
    int n0 = Nb + 64 * (wave >> 1);
    const bf16* A = n_t + (size_t)bb * HW_ * C_;
    int arow = m0 + (lane & 15);

    v8f acc[4] = {};
    for (int kk = 0; kk < C_; kk += 32) {
        v16bf a = load_frag(A, arow, C_, kk, lane);
        #pragma unroll
        for (int j = 0; j < 4; ++j) {
            v16bf bf = load_frag(wq, n0 + 16 * j + (lane & 15), C_, kk, lane);
            acc[j] = wmma_bf16(a, bf, acc[j]);
        }
    }
    int hi = lane >> 4, col = lane & 15;
    #pragma unroll
    for (int j = 0; j < 4; ++j) {
        int d = n0 + 16 * j + col;
        int h = d / 192, jj = d % 192;
        int which = jj >> 6, cc = jj & 63;
        #pragma unroll
        for (int r = 0; r < 8; ++r) {
            int p = m0 + r + 8 * hi;
            float v = acc[j][r];
            if (which == 0)
                q_t[(((size_t)bb * NH_ + h) * HW_ + p) * HD_ + cc] =
                    f2bf(v * 0.04419417382415922f);       // 1/sqrt(512)
            else if (which == 1)
                k_t[(((size_t)bb * NH_ + h) * HW_ + p) * HD_ + cc] = f2bf(v);
            else
                v_cm[(((size_t)bb * NH_ + h) * HD_ + cc) * HW_ + p] = f2bf(v);
        }
    }
}

// ---------- K3: flash attention, 16 queries/wave, 32-key tiles ----------
// Computes S^T = K.Q^T so the exp'd probabilities sit directly in the
// B-fragment layout of the PV WMMA (zero cross-lane movement).
__global__ __launch_bounds__(256) void attn_kernel(const bf16* __restrict__ q_t,
                                                   const bf16* __restrict__ k_t,
                                                   const bf16* __restrict__ v_cm,
                                                   bf16* __restrict__ o_t) {
    int bh = blockIdx.y;                 // b*8 + h
    int bb = bh >> 3, h = bh & 7;
    int tid = threadIdx.x, wave = tid >> 5, lane = tid & 31;
    int q0 = blockIdx.x * 128 + wave * 16;

    const bf16* Q = q_t + (size_t)bh * HW_ * HD_;
    const bf16* K = k_t + (size_t)bh * HW_ * HD_;
    const bf16* V = v_cm + (size_t)bh * HD_ * HW_;

    int lrow = lane & 15;
    v16bf bq0 = load_frag(Q, q0 + lrow, HD_, 0,  lane);   // B-frag: lane = query
    v16bf bq1 = load_frag(Q, q0 + lrow, HD_, 32, lane);

    v8f o0 = {}, o1 = {}, o2 = {}, o3 = {};
    float m_run = -3.0e38f, l_run = 0.f;

    for (int y0 = 0; y0 < HW_; y0 += 32) {
        // T = K_tile(32x64) . Q^T(64x16)   (two 16-row D-frags, K-dim = 64)
        v16bf a00 = load_frag(K, y0 +      lrow, HD_, 0,  lane);
        v16bf a01 = load_frag(K, y0 +      lrow, HD_, 32, lane);
        v16bf a10 = load_frag(K, y0 + 16 + lrow, HD_, 0,  lane);
        v16bf a11 = load_frag(K, y0 + 16 + lrow, HD_, 32, lane);
        v8f t0 = {}, t1 = {};
        t0 = wmma_bf16(a00, bq0, t0); t0 = wmma_bf16(a01, bq1, t0);
        t1 = wmma_bf16(a10, bq0, t1); t1 = wmma_bf16(a11, bq1, t1);

        // online softmax over the key dim (per-lane = one query column)
        float tmax = -3.0e38f;
        #pragma unroll
        for (int r = 0; r < 8; ++r) {
            tmax = fmaxf(tmax, t0[r]); tmax = fmaxf(tmax, t1[r]);
        }
        tmax = fmaxf(tmax, __shfl_xor(tmax, 16, 32));     // combine lane halves
        float m_new = fmaxf(m_run, tmax);
        float corr  = __expf(m_run - m_new);
        float p0[8], p1[8], s = 0.f;
        #pragma unroll
        for (int r = 0; r < 8; ++r) {
            p0[r] = __expf(t0[r] - m_new); s += p0[r];
            p1[r] = __expf(t1[r] - m_new); s += p1[r];
        }
        s += __shfl_xor(s, 16, 32);
        l_run = l_run * corr + s;
        m_run = m_new;
        #pragma unroll
        for (int r = 0; r < 8; ++r) {
            o0[r] *= corr; o1[r] *= corr; o2[r] *= corr; o3[r] *= corr;
        }

        // P^T in registers IS the B-fragment (32 keys x 16 queries)
        v16bf bp;
        #pragma unroll
        for (int r = 0; r < 8; ++r) { bp[r] = f2bf(p0[r]); bp[8 + r] = f2bf(p1[r]); }

        // O^T(64x16) += V_tile(64x32) . P^T(32x16)
        v16bf av0 = load_frag(V,      lrow, HW_, y0, lane);
        v16bf av1 = load_frag(V, 16 + lrow, HW_, y0, lane);
        v16bf av2 = load_frag(V, 32 + lrow, HW_, y0, lane);
        v16bf av3 = load_frag(V, 48 + lrow, HW_, y0, lane);
        o0 = wmma_bf16(av0, bp, o0);
        o1 = wmma_bf16(av1, bp, o1);
        o2 = wmma_bf16(av2, bp, o2);
        o3 = wmma_bf16(av3, bp, o3);
    }

    float inv = 1.f / l_run;
    int hi = lane >> 4;
    int p  = q0 + lrow;
    bf16* dst = o_t + ((size_t)bb * HW_ + p) * C_ + h * HD_ + 8 * hi;
    v8bf s0, s1, s2, s3;
    #pragma unroll
    for (int r = 0; r < 8; ++r) {
        s0[r] = f2bf(o0[r] * inv); s1[r] = f2bf(o1[r] * inv);
        s2[r] = f2bf(o2[r] * inv); s3[r] = f2bf(o3[r] * inv);
    }
    *reinterpret_cast<v8bf*>(dst)      = s0;
    *reinterpret_cast<v8bf*>(dst + 16) = s1;
    *reinterpret_cast<v8bf*>(dst + 32) = s2;
    *reinterpret_cast<v8bf*>(dst + 48) = s3;
}

// ---------- K4: out-proj GEMM + bias + residual ----------
__global__ __launch_bounds__(256) void proj_kernel(const bf16* __restrict__ o_t,
                                                   const bf16* __restrict__ wo,
                                                   const float* __restrict__ ob,
                                                   const float* __restrict__ x,
                                                   float* __restrict__ out) {
    int bb = blockIdx.z;
    int Mb = blockIdx.x * 32;            // position base
    int Nb = blockIdx.y * 256;           // dout base
    int tid = threadIdx.x, wave = tid >> 5, lane = tid & 31;
    int m0 = Mb + 16 * (wave & 1);
    int n0 = Nb + 64 * (wave >> 1);
    const bf16* A = o_t + (size_t)bb * HW_ * C_;
    int arow = m0 + (lane & 15);

    v8f acc[4] = {};
    for (int kk = 0; kk < C_; kk += 32) {
        v16bf a = load_frag(A, arow, C_, kk, lane);
        #pragma unroll
        for (int j = 0; j < 4; ++j) {
            v16bf bf = load_frag(wo, n0 + 16 * j + (lane & 15), C_, kk, lane);
            acc[j] = wmma_bf16(a, bf, acc[j]);
        }
    }
    int hi = lane >> 4, col = lane & 15;
    #pragma unroll
    for (int j = 0; j < 4; ++j) {
        int d = n0 + 16 * j + col;
        float bias = ob[d];
        size_t base = ((size_t)bb * C_ + d) * HW_ + m0 + 8 * hi;   // 8 consecutive p
        const float4* xr = reinterpret_cast<const float4*>(x + base);
        float4* yr       = reinterpret_cast<float4*>(out + base);
        float4 x0 = xr[0], x1 = xr[1], r0, r1;
        r0.x = acc[j][0] + bias + x0.x; r0.y = acc[j][1] + bias + x0.y;
        r0.z = acc[j][2] + bias + x0.z; r0.w = acc[j][3] + bias + x0.w;
        r1.x = acc[j][4] + bias + x1.x; r1.y = acc[j][5] + bias + x1.y;
        r1.z = acc[j][6] + bias + x1.z; r1.w = acc[j][7] + bias + x1.w;
        yr[0] = r0; yr[1] = r1;
    }
}

// ---------- host ----------
extern "C" void kernel_launch(void* const* d_in, const int* in_sizes, int n_in,
                              void* d_out, int out_size, void* d_ws, size_t ws_size,
                              hipStream_t stream) {
    const float* x     = (const float*)d_in[0];
    const float* gn_w  = (const float*)d_in[1];
    const float* gn_b  = (const float*)d_in[2];
    const float* qkv_w = (const float*)d_in[3];
    const float* out_w = (const float*)d_in[4];
    const float* out_b = (const float*)d_in[5];
    float* out = (float*)d_out;

    char* ws = (char*)d_ws;
    bf16* n_t  = (bf16*)(ws);                       //  8 MB  [B][HW][C]
    bf16* wq   = (bf16*)(ws + 8388608);             //  1.5MB [1536][512]
    bf16* wo   = (bf16*)(ws + 9961472);             //  0.5MB [512][512]
    bf16* q_t  = (bf16*)(ws + 10485760);            //  8 MB  [B][NH][HW][HD]
    bf16* k_t  = (bf16*)(ws + 18874368);            //  8 MB  [B][NH][HW][HD]
    bf16* v_cm = (bf16*)(ws + 27262976);            //  8 MB  [B][NH][HD][HW]
    bf16* o_t  = (bf16*)(ws + 35651584);            //  8 MB  [B][HW][C]

    cvt_kernel<<<(1536 * 512 + 255) / 256, 256, 0, stream>>>(qkv_w, wq, 1536 * 512);
    cvt_kernel<<<(512 * 512 + 255) / 256, 256, 0, stream>>>(out_w, wo, 512 * 512);
    gn_kernel<<<B_ * G_, 256, 0, stream>>>(x, gn_w, gn_b, n_t);
    qkv_kernel<<<dim3(HW_ / 32, 1536 / 256, B_), 256, 0, stream>>>(n_t, wq, q_t, k_t, v_cm);
    attn_kernel<<<dim3(HW_ / 128, B_ * NH_), 256, 0, stream>>>(q_t, k_t, v_cm, o_t);
    proj_kernel<<<dim3(HW_ / 32, C_ / 256, B_), 256, 0, stream>>>(o_t, wo, out_b, x, out);
}
